// HyperWeight_75041668596273
// MI455X (gfx1250) — compile-verified, matrix operand order checked by reading, not run on hardware
//
#include <hip/hip_runtime.h>
#include <hip/hip_bf16.h>

#define N_NODES 25000
#define N_EDGES 400000
#define IN_DIM  64
#define HIDDEN  128
#define RREPS   2

typedef __attribute__((ext_vector_type(16))) _Float16 v16h;
typedef __attribute__((ext_vector_type(8)))  float    v8f;

// ---------------------------------------------------------------- utilities
__global__ __launch_bounds__(256) void k_zero(float* __restrict__ p, int n) {
    int i = blockIdx.x * 256 + threadIdx.x;
    if (i < n) p[i] = 0.0f;
}

__global__ __launch_bounds__(256) void k_deg(const int* __restrict__ ei,
                                             float* __restrict__ deg) {
    int e = blockIdx.x * 256 + threadIdx.x;
    if (e < N_EDGES) {
        atomicAdd(&deg[(unsigned)ei[e]], 1.0f);
        atomicAdd(&deg[(unsigned)ei[N_EDGES + e]], 1.0f);
    }
}

__global__ __launch_bounds__(256) void k_binv(float* __restrict__ d) {
    int n = blockIdx.x * 256 + threadIdx.x;
    if (n < N_NODES) {
        float x = d[n];
        d[n] = (x > 0.0f) ? (1.0f / x) : 0.0f;
    }
}

// ------------------------------------------- layer 1: WMMA GEMM + scatter
// Block = 128 threads = 4 waves; each wave owns a 16-edge tile.
// A = [16 x 128] f16 (rep[e,0:64] ++ 0.5*(nf[u]+nf[v])), B = W1 [128 x 128],
// C = f32 [16 x 128] scattered with atomics into he[u], he[v].
__global__ __launch_bounds__(128) void k_l1_wmma(
    const int*   __restrict__ ei,    // [2, E]
    const float* __restrict__ rep,   // [E, 64]  (this rep)
    const float* __restrict__ nf,    // [N, 64]
    const float* __restrict__ W1,    // [128, 128]  k-major
    float*       __restrict__ he)    // [N, 128] accumulators
{
    __shared__ _Float16 sW[HIDDEN][136];     // transposed: sW[n][k], padded
    __shared__ _Float16 sA[4][16][136];      // per-wave A tile, padded

    const int tid  = threadIdx.x;
    const int wave = tid >> 5;
    const int lane = tid & 31;
    const int row  = lane & 15;              // edge within tile / M row
    const int half = lane >> 4;

    // Stage W1 transposed into LDS as f16.
    for (int i = tid; i < HIDDEN * HIDDEN; i += 128) {
        int k = i >> 7, n = i & 127;
        sW[n][k] = (_Float16)W1[i];          // W1[k*128 + n]
    }

    // Stage this wave's A tile (two lanes per edge; lane covers 32 columns).
    const int tileBase = blockIdx.x * 64 + wave * 16;
    {
        const int e = tileBase + row;
        const unsigned u = (unsigned)ei[e];
        const unsigned v = (unsigned)ei[N_EDGES + e];
        const float* rp = rep + (unsigned)e * IN_DIM + half * 32;
        const float* nu = nf  + u * IN_DIM + half * 32;
        const float* nv = nf  + v * IN_DIM + half * 32;
        _Float16* dst = &sA[wave][row][0];
        #pragma unroll
        for (int c = 0; c < 32; ++c) {
            dst[half * 32 + c]      = (_Float16)rp[c];
            dst[64 + half * 32 + c] = (_Float16)(0.5f * (nu[c] + nv[c]));
        }
    }
    __syncthreads();

    // A fragments: ISA 16-bit 16x32 layout.
    // lanes 0-15: row, K {0..7} U {16..23}; lanes 16-31: row, K {8..15} U {24..31}
    v16h afrag[4];
    #pragma unroll
    for (int kt = 0; kt < 4; ++kt) {
        const _Float16* ap = &sA[wave][row][kt * 32 + half * 8];
        v16h a;
        #pragma unroll
        for (int q = 0; q < 8; ++q) { a[q] = ap[q]; a[8 + q] = ap[16 + q]; }
        afrag[kt] = a;
    }

    // Scatter targets: lane holds C rows m = half*8 + j (j = VGPR index).
    // Precompute 32-bit element offsets into he so the backend can use
    // SGPR-base + 32-bit VGPR-offset addressing (no 64-bit VALU chains).
    unsigned uoff[8], voff[8];
    #pragma unroll
    for (int j = 0; j < 8; ++j) {
        int m = half * 8 + j;
        uoff[j] = (unsigned)ei[tileBase + m] * HIDDEN;
        voff[j] = (unsigned)ei[N_EDGES + tileBase + m] * HIDDEN;
    }

    const unsigned ncol = (unsigned)(lane & 15);
    #pragma unroll
    for (int nt = 0; nt < 8; ++nt) {
        v8f acc = {};
        #pragma unroll
        for (int kt = 0; kt < 4; ++kt) {
            const _Float16* bp = &sW[nt * 16 + ncol][kt * 32 + half * 8];
            v16h b;
            #pragma unroll
            for (int q = 0; q < 8; ++q) { b[q] = bp[q]; b[8 + q] = bp[16 + q]; }
            acc = __builtin_amdgcn_wmma_f32_16x16x32_f16(
                      false, afrag[kt], false, b, (short)0, acc, false, false);
        }
        const unsigned col = (unsigned)(nt * 16) + ncol;
        #pragma unroll
        for (int j = 0; j < 8; ++j) {
            atomicAdd(he + (uoff[j] + col), acc[j]);
            atomicAdd(he + (voff[j] + col), acc[j]);
        }
    }
}

// --------------- layer 1 activation + layer 2 matvec + scatter (wave/edge)
__global__ __launch_bounds__(256) void k_l2_edge(
    const int*   __restrict__ ei,
    const float* __restrict__ he,    // [N,128] raw segment sums
    const float* __restrict__ binv,  // [N]
    const float* __restrict__ b1,    // [128]
    const float* __restrict__ W2,    // [128]
    float*       __restrict__ he2)   // [N]
{
    const int wave = threadIdx.x >> 5;
    const int lane = threadIdx.x & 31;
    const int e = blockIdx.x * 8 + wave;
    if (e >= N_EDGES) return;

    const unsigned u = (unsigned)ei[e], v = (unsigned)ei[N_EDGES + e];
    const float bu = binv[u], bv = binv[v];
    const float* hu = he + (u * HIDDEN + lane * 4);
    const float* hv = he + (v * HIDDEN + lane * 4);

    float s = 0.0f;
    #pragma unroll
    for (int q = 0; q < 4; ++q) {
        float x = 0.5f * (hu[q] * bu + hv[q] * bv) + b1[lane * 4 + q];
        s += tanhf(x) * W2[lane * 4 + q];
    }
    #pragma unroll
    for (int off = 16; off > 0; off >>= 1) s += __shfl_xor(s, off, 32);
    if (lane == 0) {
        atomicAdd(he2 + u, s);
        atomicAdd(he2 + v, s);
    }
}

// ----------------------------------------------------- final edge readout
__global__ __launch_bounds__(256) void k_out(
    const int*   __restrict__ ei,
    const float* __restrict__ he2,
    const float* __restrict__ binv,
    const float* __restrict__ b2,    // [1]
    float*       __restrict__ out)   // [E] for this rep
{
    int e = blockIdx.x * 256 + threadIdx.x;
    if (e >= N_EDGES) return;
    unsigned u = (unsigned)ei[e], v = (unsigned)ei[N_EDGES + e];
    float x = 0.5f * (he2[u] * binv[u] + he2[v] * binv[v]) + b2[0];
    out[e] = 1.0f / (1.0f + expf(-x));
}

// ------------------------------------------------------------------ driver
extern "C" void kernel_launch(void* const* d_in, const int* in_sizes, int n_in,
                              void* d_out, int out_size, void* d_ws, size_t ws_size,
                              hipStream_t stream) {
    (void)in_sizes; (void)n_in; (void)out_size; (void)ws_size;

    const int*   ei  = (const int*)d_in[0];     // [2, E]
    const float* rep = (const float*)d_in[1];   // [R, E, 64]
    const float* nf  = (const float*)d_in[2];   // [N, 64]
    const float* W1  = (const float*)d_in[3];   // [128, 128]
    const float* b1  = (const float*)d_in[4];   // [128]
    const float* W2  = (const float*)d_in[5];   // [128]
    const float* b2  = (const float*)d_in[6];   // [1]
    float*       out = (float*)d_out;           // [R, E]

    // Workspace layout (256B-aligned):
    //   binv : N floats          @ 0
    //   he   : N*128 floats      @ 102400
    //   he2  : N floats          @ 102400 + N*128*4 (contiguous after he)
    float* binv = (float*)d_ws;
    float* he   = (float*)((char*)d_ws + 102400);
    float* he2  = he + (size_t)N_NODES * HIDDEN;

    const int gN  = (N_NODES + 255) / 256;
    const int gE  = (N_EDGES + 255) / 256;
    const int gHE = (N_NODES * HIDDEN + N_NODES + 255) / 256; // zero he + he2

    // Degrees -> B_inv (shared by both reps).
    k_zero<<<gN, 256, 0, stream>>>(binv, N_NODES);
    k_deg <<<gE, 256, 0, stream>>>(ei, binv);
    k_binv<<<gN, 256, 0, stream>>>(binv);

    for (int r = 0; r < RREPS; ++r) {
        const float* rep_r = rep + (size_t)r * N_EDGES * IN_DIM;
        float*       out_r = out + (size_t)r * N_EDGES;

        k_zero<<<gHE, 256, 0, stream>>>(he, N_NODES * HIDDEN + N_NODES);
        k_l1_wmma<<<N_EDGES / 64, 128, 0, stream>>>(ei, rep_r, nf, W1, he);
        k_l2_edge<<<N_EDGES / 8, 256, 0, stream>>>(ei, he, binv, b1, W2, he2);
        k_out<<<gE, 256, 0, stream>>>(ei, he2, binv, b2, out_r);
    }
}
